// SparseMoEBlock_9328668967123
// MI455X (gfx1250) — matrix operand order repaired and checked by reading, not compiled, not run on hardware
//
#include <hip/hip_runtime.h>

// ---------------- problem constants (from reference) ----------------
#define B_DIM 4
#define S_DIM 4096
#define D_DIM 1024
#define E_DIM 8
#define I_DIM 4096
#define KCAP  1024          // (S/E)*CAPACITY
#define KC    32            // WMMA K chunk (bf16 16x16x32)

typedef __attribute__((ext_vector_type(16))) __bf16 bf16x16;
typedef __attribute__((ext_vector_type(8)))  float  floatx8;

union FragU { bf16x16 v; unsigned int d[8]; uint4 q[2]; };

static __device__ __forceinline__ unsigned short f2bf(float f) {
    unsigned int u = __float_as_uint(f);
    unsigned int r = (u + 0x7FFFu + ((u >> 16) & 1u)) >> 16;
    return (unsigned short)r;
}

// ---- LDS staging: async global->LDS on CDNA5 toolchains, sync fallback ----
#if defined(__gfx1250__) && __has_builtin(__builtin_amdgcn_global_load_async_to_lds_b128)
#define MOE_ASYNC_LDS 1
#else
#define MOE_ASYNC_LDS 0
#endif

// exact pointee type from the clang diagnostic: vector_size(16) int
typedef int moe_i32x4 __attribute__((__vector_size__(4 * sizeof(int))));

static __device__ __forceinline__ void cp_lds_b128(void* ldsDst, const void* gSrc) {
#if MOE_ASYNC_LDS
    __builtin_amdgcn_global_load_async_to_lds_b128(
        (__attribute__((address_space(1))) moe_i32x4*)(unsigned long long)gSrc,
        (__attribute__((address_space(3))) moe_i32x4*)ldsDst, 0, 0);
#else
    *(uint4*)ldsDst = *(const uint4*)gSrc;
#endif
}
static __device__ __forceinline__ void cp_lds_commit_wait() {
#if MOE_ASYNC_LDS
#if __has_builtin(__builtin_amdgcn_s_wait_asynccnt)
    __builtin_amdgcn_s_wait_asynccnt(0);
#else
    asm volatile("s_wait_asynccnt 0" ::: "memory");
#endif
#endif
}

// ---------------- 0: zero output ----------------
__global__ void moe_zero_f4(float4* __restrict__ p, size_t n4) {
    size_t i = (size_t)blockIdx.x * blockDim.x + threadIdx.x;
    if (i < n4) p[i] = make_float4(0.f, 0.f, 0.f, 0.f);
}

// ---------------- 1: fp32 -> bf16 convert (4 elems / thread) ----------------
__global__ void moe_f32_to_bf16(const float4* __restrict__ src,
                                uint2* __restrict__ dst, size_t n4) {
    size_t i = (size_t)blockIdx.x * blockDim.x + threadIdx.x;
    if (i >= n4) return;
    float4 v = src[i];
    uint2 o;
    o.x = (unsigned)f2bf(v.x) | ((unsigned)f2bf(v.y) << 16);
    o.y = (unsigned)f2bf(v.z) | ((unsigned)f2bf(v.w) << 16);
    dst[i] = o;
}

// ---------------- 2: gating logits + softmax -> affinity (B,E,S) ------------
__global__ __launch_bounds__(256) void moe_gate(const float* __restrict__ x,
                                                const float* __restrict__ gate_w,
                                                float* __restrict__ aff) {
    __shared__ float gw[E_DIM * D_DIM];   // 32 KB
    const int tid = threadIdx.x;
    for (int i = tid; i < E_DIM * D_DIM; i += 256) gw[i] = gate_w[i];
    __syncthreads();

    const int token = blockIdx.x * 256 + tid;            // < B*S (exact)
    const float4* xr = (const float4*)(x + (size_t)token * D_DIM);
    float acc[E_DIM];
    #pragma unroll
    for (int e = 0; e < E_DIM; ++e) acc[e] = 0.f;

    for (int d4 = 0; d4 < D_DIM / 4; ++d4) {
        float4 xv = xr[d4];
        #pragma unroll
        for (int e = 0; e < E_DIM; ++e) {
            float4 wv = *(const float4*)&gw[e * D_DIM + d4 * 4];
            acc[e] += xv.x * wv.x + xv.y * wv.y + xv.z * wv.z + xv.w * wv.w;
        }
    }
    float mx = acc[0];
    #pragma unroll
    for (int e = 1; e < E_DIM; ++e) mx = fmaxf(mx, acc[e]);
    float s = 0.f;
    #pragma unroll
    for (int e = 0; e < E_DIM; ++e) { acc[e] = __expf(acc[e] - mx); s += acc[e]; }
    float inv = 1.0f / s;
    const int b = token >> 12;               // /S
    const int st = token & (S_DIM - 1);
    #pragma unroll
    for (int e = 0; e < E_DIM; ++e)
        aff[((size_t)(b * E_DIM + e)) * S_DIM + st] = acc[e] * inv;
}

// ---------------- 3: per-(b,e) top-k via LDS bitonic sort (descending) ------
__global__ __launch_bounds__(1024) void moe_topk(const float* __restrict__ aff,
                                                 float* __restrict__ gating,
                                                 int* __restrict__ idxw) {
    __shared__ float vals[S_DIM];
    __shared__ int   inds[S_DIM];
    const int tid = threadIdx.x;
    const int be = blockIdx.x;               // 0..31
    const float* row = aff + (size_t)be * S_DIM;

    #pragma unroll
    for (int r = 0; r < 4; ++r) {
        int j = tid + r * 1024;
        vals[j] = row[j];
        inds[j] = j;
    }
    for (unsigned ksz = 2; ksz <= S_DIM; ksz <<= 1) {
        for (unsigned j = ksz >> 1; j > 0; j >>= 1) {
            __syncthreads();
            #pragma unroll
            for (int r = 0; r < 4; ++r) {
                unsigned i = tid + r * 1024u;
                unsigned ixj = i ^ j;
                if (ixj > i) {
                    float vi = vals[i], vj = vals[ixj];
                    bool descBlock = ((i & ksz) == 0);
                    bool doswap = descBlock ? (vi < vj) : (vi > vj);
                    if (doswap) {
                        vals[i] = vj; vals[ixj] = vi;
                        int t = inds[i]; inds[i] = inds[ixj]; inds[ixj] = t;
                    }
                }
            }
        }
    }
    __syncthreads();
    if (tid < KCAP) {
        gating[(size_t)be * KCAP + tid] = vals[tid];
        idxw[(size_t)be * KCAP + tid] = inds[tid];
    }
}

// ---------------- 4: gather chosen tokens -> bf16 (B*E, KCAP, D) ------------
__global__ void moe_gather(const float* __restrict__ x,
                           const int* __restrict__ idxw,
                           unsigned short* __restrict__ xbf) {
    size_t id = (size_t)blockIdx.x * 256 + threadIdx.x;   // B*E*KCAP*(D/4)
    size_t rowi = id >> 8;                                 // /(D/4)
    int col = (int)(id & 255);
    int token = idxw[rowi];
    int b = (int)(rowi >> 13);                             // /(E*KCAP)
    float4 v = ((const float4*)x)[((size_t)b * S_DIM + token) * (D_DIM / 4) + col];
    uint2 o;
    o.x = (unsigned)f2bf(v.x) | ((unsigned)f2bf(v.y) << 16);
    o.y = (unsigned)f2bf(v.z) | ((unsigned)f2bf(v.w) << 16);
    ((uint2*)xbf)[rowi * (D_DIM / 4) + col] = o;
}

// ---- fragment builders (16-bit A 16x32 / B 32x16 layouts, ISA 7.12.2) ------
// A: lane (m=nl, half): halfs [m*KC + half*8 .. +7] and [m*KC + 16 + half*8 .. +7]
static __device__ __forceinline__ void loadFragA(FragU& f, const unsigned short* ldsA,
                                                 int m, int half) {
    f.q[0] = *(const uint4*)&ldsA[m * KC + half * 8];
    f.q[1] = *(const uint4*)&ldsA[m * KC + 16 + half * 8];
}
// B (as W rows, row n contiguous K): lane (n=nl, half): halfs [n*KC + half*8 .. +7] and [n*KC + 16 + half*8 .. +7]
static __device__ __forceinline__ void loadFragB(FragU& f, const unsigned short* ldsW,
                                                 int n, int half) {
    f.q[0] = *(const uint4*)&ldsW[n * KC + half * 8];
    f.q[1] = *(const uint4*)&ldsW[n * KC + 16 + half * 8];
}

// ---------------- 5: fused gate/up GEMM + SiLU -> h bf16 --------------------
// per (b,e): M=KCAP, N=I, K=D. Block tile 64x256, 8 waves; wave tile 32x64
// (2 m-subtiles x 4 n-subtiles; B fragments reused across both m-subtiles).
__global__ __launch_bounds__(256) void moe_gemm_gateup(
    const unsigned short* __restrict__ xbf,   // (B*E, KCAP, D)
    const unsigned short* __restrict__ wgbf,  // (E, I, D)
    const unsigned short* __restrict__ wubf,  // (E, I, D)
    unsigned short* __restrict__ h)           // (B*E, KCAP, I)
{
    __shared__ unsigned short ldsA[64 * KC];     //  4 KB
    __shared__ unsigned short ldsG[256 * KC];    // 16 KB
    __shared__ unsigned short ldsU[256 * KC];    // 16 KB

    const int tid = threadIdx.x;
    const int lane = tid & 31;
    const int w = tid >> 5;
    const int wm = w & 1;            // 2 m-groups of 32
    const int wn = w >> 1;           // 4 n-groups of 64
    const int half = lane >> 4;
    const int nl = lane & 15;

    const int n0 = blockIdx.x * 256;
    const int m0 = blockIdx.y * 64;
    const int be = blockIdx.z;
    const int e = be & (E_DIM - 1);

    const size_t xrow = (size_t)be * KCAP;
    const size_t wrow = (size_t)e * I_DIM;

    floatx8 accg[2][4] = {};
    floatx8 accu[2][4] = {};

    const int arow = tid >> 2;       // 0..63
    const int aseg = tid & 3;        // 4 x 8-half segments per 32-col row

    for (int kc = 0; kc < D_DIM; kc += KC) {
        __syncthreads();
        // stage A tile 64x32 (one b128 per thread)
        cp_lds_b128(&ldsA[arow * KC + aseg * 8],
                    xbf + (xrow + m0 + arow) * D_DIM + kc + aseg * 8);
        // stage weight tiles 256x32 each (four b128 per thread per array)
        #pragma unroll
        for (int rr = 0; rr < 4; ++rr) {
            int r = arow + rr * 64;
            cp_lds_b128(&ldsG[r * KC + aseg * 8],
                        wgbf + (wrow + n0 + r) * D_DIM + kc + aseg * 8);
            cp_lds_b128(&ldsU[r * KC + aseg * 8],
                        wubf + (wrow + n0 + r) * D_DIM + kc + aseg * 8);
        }
        cp_lds_commit_wait();
        __syncthreads();

        if (kc + KC < D_DIM) {   // gfx1250 global_prefetch for next chunk
            __builtin_prefetch(xbf + (xrow + m0 + arow) * D_DIM + kc + KC, 0, 1);
            __builtin_prefetch(wgbf + (wrow + n0 + arow) * D_DIM + kc + KC, 0, 1);
            __builtin_prefetch(wubf + (wrow + n0 + 128 + arow) * D_DIM + kc + KC, 0, 1);
        }

        FragU fa[2];
        loadFragA(fa[0], ldsA, wm * 32 + nl, half);
        loadFragA(fa[1], ldsA, wm * 32 + 16 + nl, half);

        #pragma unroll
        for (int nt = 0; nt < 4; ++nt) {
            const int n = wn * 64 + nt * 16 + nl;
            FragU fg, fu;
            loadFragB(fg, ldsG, n, half);
            loadFragB(fu, ldsU, n, half);
            #pragma unroll
            for (int mt = 0; mt < 2; ++mt) {
                accg[mt][nt] = __builtin_amdgcn_wmma_f32_16x16x32_bf16(
                    false, fa[mt].v, false, fg.v, (short)0, accg[mt][nt], false, false);
                accu[mt][nt] = __builtin_amdgcn_wmma_f32_16x16x32_bf16(
                    false, fa[mt].v, false, fu.v, (short)0, accu[mt][nt], false, false);
            }
        }
    }

    // epilogue: h = silu(g) * u   (C/D layout: m = v + half*8, n = nl)
    #pragma unroll
    for (int mt = 0; mt < 2; ++mt) {
        #pragma unroll
        for (int nt = 0; nt < 4; ++nt) {
            #pragma unroll
            for (int v = 0; v < 8; ++v) {
                float g = accg[mt][nt][v];
                float u = accu[mt][nt][v];
                float hv = (g / (1.0f + __expf(-g))) * u;
                int m = m0 + wm * 32 + mt * 16 + v + half * 8;
                int n = n0 + wn * 64 + nt * 16 + nl;
                h[((size_t)be * KCAP + m) * I_DIM + n] = f2bf(hv);
            }
        }
    }
}

// ---------------- 6: down GEMM + gating scale + scatter-add -----------------
// per (b,e): M=KCAP, N=D, K=I. Block tile 64x256, wave tile 32x64.
__global__ __launch_bounds__(256) void moe_gemm_down(
    const unsigned short* __restrict__ h,     // (B*E, KCAP, I)
    const unsigned short* __restrict__ wdbf,  // (E, D, I)
    const float* __restrict__ gating,         // (B*E, KCAP)
    const int* __restrict__ idxw,             // (B*E, KCAP)
    float* __restrict__ out)                  // (B, S, D)
{
    __shared__ unsigned short ldsA[64 * KC];     //  4 KB
    __shared__ unsigned short ldsW[256 * KC];    // 16 KB

    const int tid = threadIdx.x;
    const int lane = tid & 31;
    const int w = tid >> 5;
    const int wm = w & 1;
    const int wn = w >> 1;
    const int half = lane >> 4;
    const int nl = lane & 15;

    const int n0 = blockIdx.x * 256;
    const int m0 = blockIdx.y * 64;
    const int be = blockIdx.z;
    const int e = be & (E_DIM - 1);
    const int b = be >> 3;   // /E

    const size_t hrow = (size_t)be * KCAP;
    const size_t wrow = (size_t)e * D_DIM;

    floatx8 acc[2][4] = {};

    const int arow = tid >> 2;
    const int aseg = tid & 3;

    for (int kc = 0; kc < I_DIM; kc += KC) {
        __syncthreads();
        cp_lds_b128(&ldsA[arow * KC + aseg * 8],
                    h + (hrow + m0 + arow) * I_DIM + kc + aseg * 8);
        #pragma unroll
        for (int rr = 0; rr < 4; ++rr) {
            int r = arow + rr * 64;
            cp_lds_b128(&ldsW[r * KC + aseg * 8],
                        wdbf + (wrow + n0 + r) * I_DIM + kc + aseg * 8);
        }
        cp_lds_commit_wait();
        __syncthreads();

        if (kc + KC < I_DIM) {
            __builtin_prefetch(h + (hrow + m0 + arow) * I_DIM + kc + KC, 0, 1);
            __builtin_prefetch(wdbf + (wrow + n0 + arow) * I_DIM + kc + KC, 0, 1);
        }

        FragU fa[2];
        loadFragA(fa[0], ldsA, wm * 32 + nl, half);
        loadFragA(fa[1], ldsA, wm * 32 + 16 + nl, half);

        #pragma unroll
        for (int nt = 0; nt < 4; ++nt) {
            const int n = wn * 64 + nt * 16 + nl;
            FragU fw;
            loadFragB(fw, ldsW, n, half);
            #pragma unroll
            for (int mt = 0; mt < 2; ++mt) {
                acc[mt][nt] = __builtin_amdgcn_wmma_f32_16x16x32_bf16(
                    false, fa[mt].v, false, fw.v, (short)0, acc[mt][nt], false, false);
            }
        }
    }

    // epilogue: out[b, idx[m], n] += gating[m] * acc
    const float* gat = gating + hrow;
    const int* tok = idxw + hrow;
    #pragma unroll
    for (int mt = 0; mt < 2; ++mt) {
        #pragma unroll
        for (int v = 0; v < 8; ++v) {
            int m = m0 + wm * 32 + mt * 16 + v + half * 8;
            float gv = gat[m];
            int t = tok[m];
            float* orow = out + ((size_t)b * S_DIM + t) * D_DIM;
            #pragma unroll
            for (int nt = 0; nt < 4; ++nt) {
                int n = n0 + wn * 64 + nt * 16 + nl;
                atomicAdd(orow + n, acc[mt][nt][v] * gv);
            }
        }
    }
}

// ---------------- host-side launcher ----------------
extern "C" void kernel_launch(void* const* d_in, const int* in_sizes, int n_in,
                              void* d_out, int out_size, void* d_ws, size_t ws_size,
                              hipStream_t stream) {
    const float* x      = (const float*)d_in[0];   // (B,S,D)
    const float* gate_w = (const float*)d_in[1];   // (E,D)
    const float* w_gate = (const float*)d_in[2];   // (E,I,D)
    const float* w_up   = (const float*)d_in[3];   // (E,I,D)
    const float* w_down = (const float*)d_in[4];   // (E,D,I)
    float* out = (float*)d_out;                    // (B,S,D)

    char* ws = (char*)d_ws;
    size_t off = 0;
    auto carve = [&](size_t bytes) -> void* {
        void* p = ws + off;
        off = (off + bytes + 255) & ~(size_t)255;
        return p;
    };

    const size_t BE = (size_t)B_DIM * E_DIM;
    float*          aff    = (float*)carve(BE * S_DIM * sizeof(float));
    float*          gating = (float*)carve(BE * KCAP * sizeof(float));
    int*            idxw   = (int*)carve(BE * KCAP * sizeof(int));
    unsigned short* wg_bf  = (unsigned short*)carve((size_t)E_DIM * I_DIM * D_DIM * 2);
    unsigned short* wu_bf  = (unsigned short*)carve((size_t)E_DIM * I_DIM * D_DIM * 2);
    unsigned short* wd_bf  = (unsigned short*)carve((size_t)E_DIM * D_DIM * I_DIM * 2);
    unsigned short* x_bf   = (unsigned short*)carve(BE * KCAP * D_DIM * 2);
    unsigned short* h      = (unsigned short*)carve(BE * KCAP * I_DIM * 2);
    (void)ws_size; (void)in_sizes; (void)n_in; (void)out_size;

    // 0) zero output (scatter-add target)
    {
        size_t n4 = (size_t)B_DIM * S_DIM * D_DIM / 4;
        moe_zero_f4<<<(unsigned)((n4 + 255) / 256), 256, 0, stream>>>((float4*)out, n4);
    }
    // 1) weights fp32 -> bf16
    {
        size_t n4 = (size_t)E_DIM * I_DIM * D_DIM / 4;    // 8.39M groups
        unsigned blocks = (unsigned)((n4 + 255) / 256);
        moe_f32_to_bf16<<<blocks, 256, 0, stream>>>((const float4*)w_gate, (uint2*)wg_bf, n4);
        moe_f32_to_bf16<<<blocks, 256, 0, stream>>>((const float4*)w_up,   (uint2*)wu_bf, n4);
        moe_f32_to_bf16<<<blocks, 256, 0, stream>>>((const float4*)w_down, (uint2*)wd_bf, n4);
    }
    // 2) gating softmax -> affinity (B,E,S)
    moe_gate<<<(B_DIM * S_DIM) / 256, 256, 0, stream>>>(x, gate_w, aff);
    // 3) top-k per (b,e)
    moe_topk<<<(unsigned)BE, 1024, 0, stream>>>(aff, gating, idxw);
    // 4) gather + bf16
    {
        size_t total = BE * KCAP * (D_DIM / 4);
        moe_gather<<<(unsigned)(total / 256), 256, 0, stream>>>(x, idxw, x_bf);
    }
    // 5) fused gate/up GEMM + SiLU   (block 64x256)
    {
        dim3 grid(I_DIM / 256, KCAP / 64, (unsigned)BE);
        moe_gemm_gateup<<<grid, 256, 0, stream>>>(x_bf, wg_bf, wu_bf, h);
    }
    // 6) down GEMM + scatter-add combine   (block 64x256)
    {
        dim3 grid(D_DIM / 256, KCAP / 64, (unsigned)BE);
        moe_gemm_down<<<grid, 256, 0, stream>>>(h, wd_bf, gating, idxw, out);
    }
}
